// Decoder_62259845923338
// MI455X (gfx1250) — matrix operand order repaired and checked by reading, not compiled, not run on hardware
//
#include <hip/hip_runtime.h>
#include <hip/hip_bf16.h>
#include <math.h>

#define B_   64
#define S_   50
#define T1_  51
#define V_   30000
#define E_   256
#define H_   512
#define HE_  512
#define G3_  1536
#define HC_  1024

typedef __bf16 v16bf __attribute__((ext_vector_type(16)));
typedef float  v8f   __attribute__((ext_vector_type(8)));

union Frag { v16bf v; uint4 q[2]; };

__device__ __forceinline__ unsigned short f2bf(float f) {
  unsigned int u = __float_as_uint(f);
  u += 0x7FFFu + ((u >> 16) & 1u);           // round-to-nearest-even
  return (unsigned short)(u >> 16);
}
__device__ __forceinline__ float sigm(float x) { return 1.0f / (1.0f + expf(-x)); }

// ---------------------------------------------------------------------------
// WMMA GEMM:  C[M,N] (f32) = A[M,K] (bf16, row-major) * Wt[N,K]^T (bf16) + bias
// Block = 256 threads = 8 waves. Each wave: one 16-wide N tile x 64 M rows.
// Software-pipelined: k-chunk i+1 fragment loads are issued before the WMMAs
// consuming chunk i, so waits are partial (overlap L2 latency with matrix pipe).
// Requires: K % 64 == 0 (K in {256,512,1024} here), M % 64 == 0, N tiles of 16.
// ---------------------------------------------------------------------------
__global__ __launch_bounds__(256)
void wmma_gemm_kernel(const unsigned short* __restrict__ A,
                      const unsigned short* __restrict__ Wt,
                      float* __restrict__ C,
                      const float* __restrict__ bias,
                      int M, int N, int K) {
  const int wave   = threadIdx.x >> 5;
  const int lane   = threadIdx.x & 31;
  const int laneHi = lane >> 4;            // 0: lanes 0-15, 1: lanes 16-31
  const int l16    = lane & 15;
  const int n0     = (blockIdx.x * 8 + wave) * 16;
  if (n0 >= N) return;
  const int m0 = blockIdx.y * 64;

  v8f acc[4];
#pragma unroll
  for (int i = 0; i < 4; ++i)
#pragma unroll
    for (int j = 0; j < 8; ++j) acc[i][j] = 0.0f;

  // B fragment: lane holds col n, K chunks k0+{0,16}+(laneHi?8:0)
  const unsigned short* bptr = Wt + (size_t)(n0 + l16) * K + laneHi * 8;
  // A fragment: lane holds row m, K chunks k0+{0,16}+(laneHi?8:0)
  const unsigned short* aptr = A + (size_t)(m0 + l16) * K + laneHi * 8;

  Frag b0, b1;
  Frag a0[4], a1[4];

  auto loadB = [&](Frag& f, int k0) {
    f.q[0] = *(const uint4*)(bptr + k0);
    f.q[1] = *(const uint4*)(bptr + k0 + 16);
  };
  auto loadA = [&](Frag* f, int k0) {
#pragma unroll
    for (int mt = 0; mt < 4; ++mt) {
      const unsigned short* ap = aptr + (size_t)mt * 16 * K + k0;
      f[mt].q[0] = *(const uint4*)(ap);
      f[mt].q[1] = *(const uint4*)(ap + 16);
    }
  };
  auto mma4 = [&](Frag* f, Frag& b) {
#pragma unroll
    for (int mt = 0; mt < 4; ++mt)
      acc[mt] = __builtin_amdgcn_wmma_f32_16x16x32_bf16(
          false, f[mt].v, false, b.v, (short)0, acc[mt], false, false);
  };

  loadB(b0, 0);
  loadA(a0, 0);
  for (int k0 = 0; k0 < K; k0 += 64) {
    loadB(b1, k0 + 32);              // prefetch chunk i+1
    loadA(a1, k0 + 32);
    mma4(a0, b0);                    // compute chunk i
    if (k0 + 64 < K) {
      loadB(b0, k0 + 64);            // prefetch chunk i+2
      loadA(a0, k0 + 64);
    }
    mma4(a1, b1);                    // compute chunk i+1
  }

  const float bv = bias ? bias[n0 + l16] : 0.0f;
#pragma unroll
  for (int mt = 0; mt < 4; ++mt) {
    const int mbase = m0 + mt * 16 + laneHi * 8;
#pragma unroll
    for (int r = 0; r < 8; ++r)
      C[(size_t)(mbase + r) * N + n0 + l16] = acc[mt][r] + bv;
  }
}

// --------------------------- setup / elementwise ---------------------------
__global__ __launch_bounds__(256)
void cvt_kernel(const float* __restrict__ src, unsigned short* __restrict__ dst, int n) {
  int i = blockIdx.x * 256 + threadIdx.x;
  if (i < n) dst[i] = f2bf(src[i]);
}

// src f32 [K,N] -> dst bf16 [N,K]
__global__ __launch_bounds__(256)
void cvt_t_kernel(const float* __restrict__ src, unsigned short* __restrict__ dst,
                  int K, int N) {
  int i = blockIdx.x * 256 + threadIdx.x;
  if (i < K * N) {
    int n = i / K, k = i - n * K;
    dst[i] = f2bf(src[(size_t)k * N + n]);
  }
}

__global__ __launch_bounds__(256)
void inith_kernel(const float* __restrict__ eh, float* __restrict__ h,
                  unsigned short* __restrict__ hb) {
  int i = blockIdx.x * 256 + threadIdx.x;
  if (i < B_ * H_) { float v = eh[i]; h[i] = v; hb[i] = f2bf(v); }
}

__global__ __launch_bounds__(256)
void hout_kernel(const float* __restrict__ h, float* __restrict__ out) {
  int i = blockIdx.x * 256 + threadIdx.x;
  if (i < B_ * H_) out[(size_t)B_ * T1_ * V_ + i] = h[i];
}

__global__ __launch_bounds__(256)
void embed_kernel(const int* __restrict__ target, const float* __restrict__ emb,
                  unsigned short* __restrict__ xb, int t) {
  int i = blockIdx.x * 256 + threadIdx.x;      // B_*E_ threads
  int b = i >> 8, e = i & (E_ - 1);
  int tok = (t == 0) ? 1 : target[b * T1_ + t - 1];   // SOS==1
  xb[i] = f2bf(emb[(size_t)tok * E_ + e]);
}

__global__ __launch_bounds__(256)
void gru_gate_kernel(const float* __restrict__ gi, const float* __restrict__ gh,
                     float* __restrict__ h, unsigned short* __restrict__ hb) {
  int i = blockIdx.x * 256 + threadIdx.x;      // B_*H_ threads
  int b = i >> 9, j = i & (H_ - 1);
  const float* gib = gi + (size_t)b * G3_;
  const float* ghb = gh + (size_t)b * G3_;
  float r = sigm(gib[j] + ghb[j]);
  float z = sigm(gib[H_ + j] + ghb[H_ + j]);
  float n = tanhf(gib[2 * H_ + j] + r * ghb[2 * H_ + j]);
  float hn = (1.0f - z) * n + z * h[i];
  h[i] = hn;
  hb[i] = f2bf(hn);
}

__global__ __launch_bounds__(256)
void tanh_bf16_kernel(const float* __restrict__ src, unsigned short* __restrict__ dst, int n) {
  int i = blockIdx.x * 256 + threadIdx.x;
  if (i < n) dst[i] = f2bf(tanhf(src[i]));
}

// ---------------- attention: scores -> softmax -> ctx -> [h,ctx] bf16 -------
__global__ __launch_bounds__(256)
void attn_kernel(const float* __restrict__ enc_proj, const float* __restrict__ hWd,
                 const float* __restrict__ va, const float* __restrict__ enc_out,
                 const float* __restrict__ h, unsigned short* __restrict__ hc) {
  const int b = blockIdx.x;
  __shared__ float sc[S_];
  __shared__ float att[S_];
  const int wave = threadIdx.x >> 5, lane = threadIdx.x & 31;

  for (int s = wave; s < S_; s += 8) {
    float p = 0.0f;
    const float* ep = enc_proj + ((size_t)b * S_ + s) * H_;
    const float* hw = hWd + (size_t)b * H_;
    for (int k = lane; k < H_; k += 32) p += tanhf(ep[k] + hw[k]) * va[k];
    for (int off = 16; off; off >>= 1) p += __shfl_xor(p, off, 32);
    if (lane == 0) sc[s] = p;
  }
  __syncthreads();

  float m = -1e30f;
  for (int s = 0; s < S_; ++s) m = fmaxf(m, sc[s]);
  float tot = 0.0f;
  for (int s = 0; s < S_; ++s) tot += expf(sc[s] - m);
  if (threadIdx.x < S_) att[threadIdx.x] = expf(sc[threadIdx.x] - m) / tot;
  __syncthreads();

  for (int e = threadIdx.x; e < HE_; e += 256) {
    float c = 0.0f;
    for (int s = 0; s < S_; ++s)
      c += att[s] * enc_out[((size_t)b * S_ + s) * HE_ + e];
    unsigned short* row = hc + (size_t)b * HC_;
    row[e]       = f2bf(h[(size_t)b * H_ + e]);   // H_ == HE_ == 512
    row[H_ + e]  = f2bf(c);
  }
}

// ----------------------------- log_softmax over V ---------------------------
__global__ __launch_bounds__(256)
void logsoftmax_kernel(const float* __restrict__ logits, float* __restrict__ out, int t) {
  const int b = blockIdx.x;
  const float* row = logits + (size_t)b * V_;
  __shared__ float red[8];

  float m = -1e30f;
  for (int v = threadIdx.x; v < V_; v += 256) m = fmaxf(m, row[v]);
  for (int off = 16; off; off >>= 1) m = fmaxf(m, __shfl_xor(m, off, 32));
  if ((threadIdx.x & 31) == 0) red[threadIdx.x >> 5] = m;
  __syncthreads();
  m = red[0];
#pragma unroll
  for (int i = 1; i < 8; ++i) m = fmaxf(m, red[i]);
  __syncthreads();

  float s = 0.0f;
  for (int v = threadIdx.x; v < V_; v += 256) s += expf(row[v] - m);
  for (int off = 16; off; off >>= 1) s += __shfl_xor(s, off, 32);
  if ((threadIdx.x & 31) == 0) red[threadIdx.x >> 5] = s;
  __syncthreads();
  s = 0.0f;
#pragma unroll
  for (int i = 0; i < 8; ++i) s += red[i];

  const float lse = m + logf(s);
  float* orow = out + (size_t)b * T1_ * V_ + (size_t)t * V_;
  for (int v = threadIdx.x; v < V_; v += 256) orow[v] = row[v] - lse;
}

// ---------------------------------------------------------------------------
extern "C" void kernel_launch(void* const* d_in, const int* in_sizes, int n_in,
                              void* d_out, int out_size, void* d_ws, size_t ws_size,
                              hipStream_t stream) {
  const int*   target  = (const int*)  d_in[0];
  const float* enc_hid = (const float*)d_in[1];
  const float* enc_out = (const float*)d_in[2];
  const float* emb     = (const float*)d_in[3];
  const float* W_ih    = (const float*)d_in[4];
  const float* W_hh    = (const float*)d_in[5];
  const float* b_ih    = (const float*)d_in[6];
  const float* b_hh    = (const float*)d_in[7];
  const float* Wd      = (const float*)d_in[8];
  const float* We      = (const float*)d_in[9];
  const float* va      = (const float*)d_in[10];
  const float* Wa      = (const float*)d_in[11];
  const float* fc_w    = (const float*)d_in[12];
  const float* fc_b    = (const float*)d_in[13];
  float* out = (float*)d_out;

  // ---- workspace carve-up (bf16 stored as ushort) ----
  char* p = (char*)d_ws;
  auto alloc = [&](size_t bytes) {
    char* r = p; p += (bytes + 255) & ~(size_t)255; return r;
  };
  unsigned short* Wih_b  = (unsigned short*)alloc((size_t)G3_ * E_ * 2);   // [1536,256] already [N,K]
  unsigned short* Whh_b  = (unsigned short*)alloc((size_t)G3_ * H_ * 2);   // [1536,512]
  unsigned short* Wd_t   = (unsigned short*)alloc((size_t)H_  * H_ * 2);   // [512,512]
  unsigned short* We_t   = (unsigned short*)alloc((size_t)H_  * HE_ * 2);  // [512,512]
  unsigned short* Wa_t   = (unsigned short*)alloc((size_t)H_  * HC_ * 2);  // [512,1024]
  unsigned short* fcw_t  = (unsigned short*)alloc((size_t)V_  * H_ * 2);   // [30000,512]
  unsigned short* encb   = (unsigned short*)alloc((size_t)B_ * S_ * HE_ * 2);
  float*          encprj = (float*)alloc((size_t)B_ * S_ * H_ * 4);
  float*          h_f    = (float*)alloc((size_t)B_ * H_ * 4);
  unsigned short* h_b    = (unsigned short*)alloc((size_t)B_ * H_ * 2);
  unsigned short* x_b    = (unsigned short*)alloc((size_t)B_ * E_ * 2);
  float*          gi     = (float*)alloc((size_t)B_ * G3_ * 4);
  float*          gh     = (float*)alloc((size_t)B_ * G3_ * 4);
  float*          hWd    = (float*)alloc((size_t)B_ * H_ * 4);
  unsigned short* hc_b   = (unsigned short*)alloc((size_t)B_ * HC_ * 2);
  float*          o_pre  = (float*)alloc((size_t)B_ * H_ * 4);
  unsigned short* o_b    = (unsigned short*)alloc((size_t)B_ * H_ * 2);
  float*          logits = (float*)alloc((size_t)B_ * V_ * 4);

  auto blocks = [](long n) { return (unsigned)((n + 255) / 256); };

  // ---- one-time weight conversion (in-graph; deterministic) ----
  cvt_kernel<<<blocks((long)G3_ * E_), 256, 0, stream>>>(W_ih, Wih_b, G3_ * E_);
  cvt_kernel<<<blocks((long)G3_ * H_), 256, 0, stream>>>(W_hh, Whh_b, G3_ * H_);
  cvt_t_kernel<<<blocks((long)H_ * H_), 256, 0, stream>>>(Wd, Wd_t, H_, H_);
  cvt_t_kernel<<<blocks((long)HE_ * H_), 256, 0, stream>>>(We, We_t, HE_, H_);
  cvt_t_kernel<<<blocks((long)HC_ * H_), 256, 0, stream>>>(Wa, Wa_t, HC_, H_);
  cvt_t_kernel<<<blocks((long)H_ * V_), 256, 0, stream>>>(fc_w, fcw_t, H_, V_);
  cvt_kernel<<<blocks((long)B_ * S_ * HE_), 256, 0, stream>>>(enc_out, encb, B_ * S_ * HE_);
  inith_kernel<<<blocks((long)B_ * H_), 256, 0, stream>>>(enc_hid, h_f, h_b);

  // enc_proj = enc_out @ We : [3200,512] x [512,512]
  {
    dim3 g((H_ + 127) / 128, (B_ * S_) / 64);
    wmma_gemm_kernel<<<g, 256, 0, stream>>>(encb, We_t, encprj, nullptr,
                                            B_ * S_, H_, HE_);
  }

  // ---- 51 teacher-forced decoder steps ----
  for (int t = 0; t < T1_; ++t) {
    embed_kernel<<<blocks((long)B_ * E_), 256, 0, stream>>>(target, emb, x_b, t);

    dim3 g_gru((G3_ + 127) / 128, 1);
    wmma_gemm_kernel<<<g_gru, 256, 0, stream>>>(x_b, Wih_b, gi, b_ih, B_, G3_, E_);
    wmma_gemm_kernel<<<g_gru, 256, 0, stream>>>(h_b, Whh_b, gh, b_hh, B_, G3_, H_);
    gru_gate_kernel<<<blocks((long)B_ * H_), 256, 0, stream>>>(gi, gh, h_f, h_b);

    dim3 g_h((H_ + 127) / 128, 1);
    wmma_gemm_kernel<<<g_h, 256, 0, stream>>>(h_b, Wd_t, hWd, nullptr, B_, H_, H_);
    attn_kernel<<<B_, 256, 0, stream>>>(encprj, hWd, va, enc_out, h_f, hc_b);

    wmma_gemm_kernel<<<g_h, 256, 0, stream>>>(hc_b, Wa_t, o_pre, nullptr, B_, H_, HC_);
    tanh_bf16_kernel<<<blocks((long)B_ * H_), 256, 0, stream>>>(o_pre, o_b, B_ * H_);

    dim3 g_fc((V_ + 127) / 128, 1);
    wmma_gemm_kernel<<<g_fc, 256, 0, stream>>>(o_b, fcw_t, logits, fc_b, B_, V_, H_);
    logsoftmax_kernel<<<B_, 256, 0, stream>>>(logits, out, t);
  }

  hout_kernel<<<blocks((long)B_ * H_), 256, 0, stream>>>(h_f, out);
}
